// AddPathDecoder_12120397710139
// MI455X (gfx1250) — matrix operand order repaired and checked by reading, not compiled, not run on hardware
//
#include <hip/hip_runtime.h>
#include <hip/hip_bf16.h>
#include <math.h>

#define B_  64
#define N_  2000
#define G_  128
#define H_  128
#define NH_ 8
#define NP_ 2016      // padded Vt row length (N rounded up to 32) so tail A-frag loads stay in-row
#define NSPLIT 5
#define TILES_PER 25  // 125 n-tiles / 5 splits
#define GB_ 16        // G/8 mask bytes per (b,n)

typedef __bf16 bf16;
typedef __attribute__((ext_vector_type(16))) bf16  bf16x16;
typedef __attribute__((ext_vector_type(8)))  bf16  bf16x8;
typedef __attribute__((ext_vector_type(8)))  float f32x8;
typedef __attribute__((ext_vector_type(4)))  float f32x4;

// ---------------- workspace layout (bytes) ----------------
#define OFF_EMB   0u                 // bf16 (B,N,H)      32,768,000
#define OFF_K     32768000u          // bf16 (B,N,H)      32,768,000
#define OFF_VT    65536000u          // bf16 (B,H,NP)     33,030,144
#define OFF_WK    98566144u          // bf16 128x128
#define OFF_WV    98598912u
#define OFF_WQL   98631680u
#define OFF_WC    98664448u
#define OFF_MEAN  98697216u          // f32 (B,H)
#define OFF_QB    98729984u          // f32 (B,H)  combined graph+src+tgt query
#define OFF_Q     98762752u          // bf16 (B,G,H) glimpse_q * 0.25
#define OFF_MH    100859904u         // f32 (B,G,H) multi-head out
#define OFF_FQ    105054208u         // bf16 (B,G,H) final_q / sqrt(H)
#define OFF_RS    107151360u         // f32 (B,G) reciprocal row sums
#define OFF_RSP   107184128u         // f32 (B,G,NSPLIT) partial row sums
#define OFF_MP    107347968u         // u8  (B,N,G/8) packed mask bits   2,048,000

// ---------------- helpers ----------------
static __device__ __forceinline__ f32x8 wmma_bf(bf16x16 a, bf16x16 b, f32x8 c) {
  return __builtin_amdgcn_wmma_f32_16x16x32_bf16(false, a, false, b, (short)0, c, false, false);
}
static __device__ __forceinline__ bf16x8  ldb8 (const bf16* p){ return *(const bf16x8 *)p; }
static __device__ __forceinline__ bf16x16 ldb16(const bf16* p){ return *(const bf16x16*)p; }
static __device__ __forceinline__ bf16x16 catb(bf16x8 lo, bf16x8 hi){
  bf16x16 r;
#pragma unroll
  for (int i=0;i<8;i++){ r[i]=lo[i]; r[i+8]=hi[i]; }
  return r;
}
static __device__ __forceinline__ bf16x8 zb8(){ bf16x8 r;
#pragma unroll
  for (int i=0;i<8;i++) r[i]=(bf16)0.f; return r; }
static __device__ __forceinline__ f32x8 zf8(){ f32x8 r;
#pragma unroll
  for (int i=0;i<8;i++) r[i]=0.f; return r; }
static __device__ __forceinline__ bf16x8 cvt8(const float* p){ bf16x8 r;
#pragma unroll
  for (int i=0;i<8;i++) r[i]=(bf16)p[i]; return r; }
// branch-free tanh: 1 - 2/(e^{2x}+1); saturates to +/-1 correctly as e^{2x} -> inf / 0
static __device__ __forceinline__ float tanh_fast(float x){
  return 1.0f - 2.0f/(__expf(2.0f*x) + 1.0f);
}

// ---------------- 1. conversions ----------------
__global__ void k_cvt_emb(const float* __restrict__ src, bf16* __restrict__ dst){
  size_t i = (size_t)blockIdx.x*256 + threadIdx.x;
  dst[i] = (bf16)src[i];
}
__global__ void k_cvt_w(const float* wk, const float* wv, const float* wql, const float* wc,
                        bf16* ok, bf16* ov, bf16* oql, bf16* oc){
  int i = blockIdx.x*256 + threadIdx.x;       // 4*16384
  int m = i >> 14, j = i & 16383;
  if      (m==0) ok [j] = (bf16)wk [j];
  else if (m==1) ov [j] = (bf16)wv [j];
  else if (m==2) oql[j] = (bf16)wql[j];
  else           oc [j] = (bf16)wc [j];
}

// ---------------- 1b. pack mask (0 / -1e9) into bits: mp[b][n][g/8], bit r -> g = gb*8+r ----------------
__global__ void k_mpack(const float* __restrict__ mask, unsigned char* __restrict__ mp){
  int t = blockIdx.x*256 + threadIdx.x;       // B * GB * N  (n fastest for coalescing)
  int n  = t % N_;
  int bg = t / N_;                            // b*GB + gb
  int b  = bg >> 4, gb = bg & 15;
  const float* m0 = mask + ((size_t)b*G_ + gb*8)*N_ + n;
  unsigned byte = 0;
#pragma unroll
  for (int j=0;j<8;j++) byte |= (m0[(size_t)j*N_] >= -0.5f) ? (1u<<j) : 0u;
  mp[((size_t)b*N_ + n)*GB_ + gb] = (unsigned char)byte;
}

// ---------------- 2. mean over N ----------------
__global__ void k_mean(const float* __restrict__ emb, float* __restrict__ gm){
  int t = blockIdx.x*256 + threadIdx.x;       // B*H
  int b = t >> 7, h = t & 127;
  const float* p = emb + (size_t)b*N_*H_ + h;
  float s = 0.f;
  for (int n=0; n<N_; ++n) s += p[(size_t)n*H_];
  gm[t] = s * (1.0f/N_);
}

// ---------------- 3. q base = Wqg*mean + Wqs*src + Wqt*tgt ----------------
__global__ void k_qbase(const float* __restrict__ gm, const float* __restrict__ emb,
                        const float* __restrict__ Wg, const float* __restrict__ Ws,
                        const float* __restrict__ Wt,
                        const int* __restrict__ srcn, const int* __restrict__ tgtn,
                        float* __restrict__ qb){
  int t = blockIdx.x*256 + threadIdx.x;       // B*H
  int b = t >> 7, h = t & 127;
  const float* ge = gm  + (size_t)b*H_;
  const float* se = emb + ((size_t)b*N_ + srcn[b])*H_;
  const float* te = emb + ((size_t)b*N_ + tgtn[b])*H_;
  const float* wg = Wg + (size_t)h*H_;
  const float* ws = Ws + (size_t)h*H_;
  const float* wt = Wt + (size_t)h*H_;
  float s = 0.f;
#pragma unroll 4
  for (int k=0;k<H_;++k) s += ge[k]*wg[k] + se[k]*ws[k] + te[k]*wt[k];
  qb[t] = s;
}

// ---------------- 4. K projection: D(c,n)=sum_k W[c,k]*emb[n,k]; store K row-major ----------------
__global__ void k_kproj(const bf16* __restrict__ emb, const bf16* __restrict__ W,
                        bf16* __restrict__ Kb){
  int gw   = blockIdx.x*8 + (threadIdx.x>>5);   // 8000 waves: (b, n-tile)
  int lane = threadIdx.x & 31, lq = lane & 15, half = lane >> 4;
  int b = gw / 125, nt = gw % 125, n0 = nt*16;
  const bf16* erow = emb + ((size_t)b*N_ + n0 + lq)*H_;
  bf16x16 bfrag[4];
#pragma unroll
  for (int kk=0;kk<4;kk++) bfrag[kk] = ldb16(erow + kk*32 + half*16);
#pragma unroll
  for (int ct=0;ct<8;ct++){
    f32x8 acc = zf8();
#pragma unroll
    for (int kk=0;kk<4;kk++){
      const bf16* wrow = W + (size_t)(ct*16 + lq)*H_ + kk*32;
      bf16x16 a = ldb16(wrow + half*16);
      acc = wmma_bf(a, bfrag[kk], acc);
    }
    bf16x8 s;
#pragma unroll
    for (int r=0;r<8;r++) s[r] = (bf16)acc[r];
    *(bf16x8*)(Kb + ((size_t)b*N_ + n0 + lq)*H_ + ct*16 + 8*half) = s;
  }
}

// ---------------- 5. V projection: D(n,c)=sum_k emb[n,k]*W[c,k]; store Vt (B,H,NP) ----------------
__global__ void k_vproj(const bf16* __restrict__ emb, const bf16* __restrict__ W,
                        bf16* __restrict__ Vt){
  int gw   = blockIdx.x*8 + (threadIdx.x>>5);   // 8000 waves
  int lane = threadIdx.x & 31, lq = lane & 15, half = lane >> 4;
  int b = gw / 125, nt = gw % 125, n0 = nt*16;
  const bf16* erow = emb + ((size_t)b*N_ + n0 + lq)*H_;
  bf16x16 afrag[4];
#pragma unroll
  for (int kk=0;kk<4;kk++)
    afrag[kk] = ldb16(erow + kk*32 + half*16);
#pragma unroll
  for (int ct=0;ct<8;ct++){
    f32x8 acc = zf8();
#pragma unroll
    for (int kk=0;kk<4;kk++){
      bf16x16 bfr = ldb16(W + (size_t)(ct*16 + lq)*H_ + kk*32 + half*16);
      acc = wmma_bf(afrag[kk], bfr, acc);
    }
    bf16x8 s;
#pragma unroll
    for (int r=0;r<8;r++) s[r] = (bf16)acc[r];
    *(bf16x8*)(Vt + ((size_t)b*H_ + ct*16 + lq)*NP_ + n0 + 8*half) = s;
  }
}

// ---------------- 6. glimpse_q = (qbase + emb[last]@Wql^T) * 0.25, bf16 (B,G,H) ----------------
__global__ void k_qlast(const bf16* __restrict__ emb, const bf16* __restrict__ W,
                        const float* __restrict__ qb, const int* __restrict__ lastn,
                        bf16* __restrict__ Q){
  int gw   = blockIdx.x*8 + (threadIdx.x>>5);   // 512 waves: (b, g-tile)
  int lane = threadIdx.x & 31, lq = lane & 15, half = lane >> 4;
  int b = gw >> 3, gt = gw & 7, g0 = gt*16;
  int idx = lastn[b*G_ + g0 + lq];
  const bf16* erow = emb + ((size_t)b*N_ + idx)*H_;
  bf16x16 afrag[4];
#pragma unroll
  for (int kk=0;kk<4;kk++)
    afrag[kk] = ldb16(erow + kk*32 + half*16);
#pragma unroll
  for (int ct=0;ct<8;ct++){
    f32x8 acc = zf8();
#pragma unroll
    for (int kk=0;kk<4;kk++){
      bf16x16 bfr = ldb16(W + (size_t)(ct*16 + lq)*H_ + kk*32 + half*16);
      acc = wmma_bf(afrag[kk], bfr, acc);
    }
    float base = qb[(size_t)b*H_ + ct*16 + lq];
#pragma unroll
    for (int r=0;r<8;r++){
      float v = (acc[r] + base) * 0.25f;                  // fold 1/sqrt(hd)
      Q[((size_t)b*G_ + g0 + r + 8*half)*H_ + ct*16 + lq] = (bf16)v;
    }
  }
}

// ---------------- 7. attention: streaming softmax, Oᵀ = Vᵀ · attnᵀ ----------------
__global__ void k_attn(const bf16* __restrict__ Q, const bf16* __restrict__ Kb,
                       const bf16* __restrict__ Vt, const unsigned char* __restrict__ mp,
                       float* __restrict__ mh){
  __shared__ __align__(64) bf16 attn_t[4][16][32];
  __shared__ float denom[4][16];
  int w    = threadIdx.x >> 5;
  int gw   = blockIdx.x*4 + w;                 // 4096 waves: (b, h, g-tile)
  int lane = threadIdx.x & 31, lq = lane & 15, half = lane >> 4;
  int b = gw >> 6, rem = gw & 63, h = rem >> 3, gt = rem & 7, g0 = gt*16;
  int gb = gt*2 + half;                        // mask byte index: bits are g = g0+8*half+0..7

  // A-frag of q (16g x 32, upper K zero-padded; hd=16)
  const bf16* qrow = Q + ((size_t)b*G_ + g0 + lq)*H_ + h*16;
  bf16x16 aq = half ? catb(zb8(), zb8()) : ldb16(qrow);

  f32x8 o = zf8();
  float lacc[8];
#pragma unroll
  for (int r=0;r<8;r++) lacc[r]=0.f;

  const unsigned char* mrow = mp + (size_t)b*N_*GB_ + (size_t)lq*GB_ + gb;
  const bf16* vrow0 = Vt + ((size_t)b*H_ + h*16 + lq)*NP_;

  for (int c=0;c<63;++c){
    int nbase = c*32;
#pragma unroll
    for (int t=0;t<2;t++){
      int n0 = nbase + t*16;
      float ev[8];
      if (n0 < N_){
        // all lanes load; lanes 16-31 feed the zero-padded K half (result unaffected)
        bf16x16 bk = ldb16(Kb + ((size_t)b*N_ + n0 + lq)*H_ + h*16);
        f32x8 s = wmma_bf(aq, bk, zf8());
        unsigned mbyte = mrow[(size_t)n0*GB_];
#pragma unroll
        for (int r=0;r<8;r++) ev[r] = (mbyte & (1u<<r)) ? __expf(s[r]) : 0.f;
      } else {
#pragma unroll
        for (int r=0;r<8;r++) ev[r] = 0.f;
      }
#pragma unroll
      for (int r=0;r<8;r++){
        lacc[r] += ev[r];
        attn_t[w][r + 8*half][t*16 + lq] = (bf16)ev[r];   // attn[g][n] row-major
      }
    }
    // heads update: A = Vt tile (16d x 32n), B = attnᵀ from LDS
    const bf16* vrow = vrow0 + nbase;
    bf16x16 av = ldb16(vrow + half*16);
    bf16x16 ab = *(const bf16x16*)&attn_t[w][lq][half*16];
    o = wmma_bf(av, ab, o);
  }
  // row sums (per g) : reduce over the 16 lanes of each half
#pragma unroll
  for (int r=0;r<8;r++){
    float s = lacc[r];
    s += __shfl_xor(s,1); s += __shfl_xor(s,2); s += __shfl_xor(s,4); s += __shfl_xor(s,8);
    if (lq == 0) denom[w][r + 8*half] = s;
  }
  float inv = 1.0f / denom[w][lq];              // Oᵀ column g = lq
  float* orow = mh + ((size_t)b*G_ + g0 + lq)*H_ + h*16 + 8*half;
  f32x4 s0, s1;
#pragma unroll
  for (int r=0;r<4;r++){ s0[r] = o[r]*inv; s1[r] = o[r+4]*inv; }
  *(f32x4*)orow       = s0;
  *(f32x4*)(orow + 4) = s1;
}

// ---------------- 8. final_q = (mh@Wc^T + bc)/sqrt(H), bf16 ----------------
__global__ void k_fq(const float* __restrict__ mh, const bf16* __restrict__ Wc,
                     const float* __restrict__ bc, bf16* __restrict__ FQ){
  int gw   = blockIdx.x*8 + (threadIdx.x>>5);   // 512 waves
  int lane = threadIdx.x & 31, lq = lane & 15, half = lane >> 4;
  int b = gw >> 3, gt = gw & 7, g0 = gt*16;
  const float* mrow = mh + ((size_t)b*G_ + g0 + lq)*H_;
  bf16x16 afrag[4];
#pragma unroll
  for (int kk=0;kk<4;kk++)
    afrag[kk] = catb(cvt8(mrow + kk*32 + half*16), cvt8(mrow + kk*32 + half*16 + 8));
#pragma unroll
  for (int ct=0;ct<8;ct++){
    f32x8 acc = zf8();
#pragma unroll
    for (int kk=0;kk<4;kk++){
      bf16x16 bfr = ldb16(Wc + (size_t)(ct*16 + lq)*H_ + kk*32 + half*16);
      acc = wmma_bf(afrag[kk], bfr, acc);
    }
    float bias = bc[ct*16 + lq];
#pragma unroll
    for (int r=0;r<8;r++){
      float v = (acc[r] + bias) * 0.08838834764831843f;   // 1/sqrt(128)
      FQ[((size_t)b*G_ + g0 + r + 8*half)*H_ + ct*16 + lq] = (bf16)v;
    }
  }
}

// ---------------- 9. logits + clip-tanh + exp; deterministic split row sums ----------------
__global__ void k_logits(const bf16* __restrict__ FQ, const bf16* __restrict__ emb,
                         const unsigned char* __restrict__ mp, float* __restrict__ out,
                         float* __restrict__ part){
  int gw   = blockIdx.x*8 + (threadIdx.x>>5);   // 2560 waves: (b, g-tile, split)
  int lane = threadIdx.x & 31, lq = lane & 15, half = lane >> 4;
  int b = gw / (8*NSPLIT), rem = gw % (8*NSPLIT), gt = rem / NSPLIT, sp = rem % NSPLIT;
  int g0 = gt*16, nt0 = sp*TILES_PER;
  int gb = gt*2 + half;
  const bf16* fqrow = FQ + ((size_t)b*G_ + g0 + lq)*H_;
  bf16x16 afrag[4];
#pragma unroll
  for (int kk=0;kk<4;kk++)
    afrag[kk] = ldb16(fqrow + kk*32 + half*16);

  float lacc[8];
#pragma unroll
  for (int r=0;r<8;r++) lacc[r]=0.f;
  const unsigned char* mrow = mp + (size_t)b*N_*GB_ + (size_t)lq*GB_ + gb;
  float* obase = out + ((size_t)b*G_ + g0 + 8*half)*N_ + lq;

#pragma unroll 2
  for (int nt=0; nt<TILES_PER; ++nt){
    int n0 = (nt0 + nt)*16;
    f32x8 acc = zf8();
#pragma unroll
    for (int kk=0;kk<4;kk++){
      bf16x16 bfr = ldb16(emb + ((size_t)b*N_ + n0 + lq)*H_ + kk*32 + half*16);
      acc = wmma_bf(afrag[kk], bfr, acc);
    }
    unsigned mbyte = mrow[(size_t)n0*GB_];
#pragma unroll
    for (int r=0;r<8;r++){
      float e = (mbyte & (1u<<r)) ? __expf(10.0f*tanh_fast(acc[r])) : 0.f;
      lacc[r] += e;
      obase[(size_t)r*N_ + n0] = e;
    }
  }
#pragma unroll
  for (int r=0;r<8;r++){
    float s = lacc[r];
    s += __shfl_xor(s,1); s += __shfl_xor(s,2); s += __shfl_xor(s,4); s += __shfl_xor(s,8);
    if (lq == 0) part[(size_t)(b*G_ + g0 + r + 8*half)*NSPLIT + sp] = s;
  }
}

__global__ void k_sumpart(const float* __restrict__ part, float* __restrict__ rs){
  int t = blockIdx.x*256 + threadIdx.x;         // B*G
  float s = 0.f;
#pragma unroll
  for (int i=0;i<NSPLIT;i++) s += part[(size_t)t*NSPLIT + i];
  rs[t] = 1.0f / s;                             // store reciprocal: k_norm multiplies
}

__global__ void k_norm(float* __restrict__ out, const float* __restrict__ rs){
  size_t i = (size_t)blockIdx.x*256 + threadIdx.x;   // B*G*N
  out[i] = out[i] * rs[i / N_];
}

// ---------------- launch ----------------
extern "C" void kernel_launch(void* const* d_in, const int* in_sizes, int n_in,
                              void* d_out, int out_size, void* d_ws, size_t ws_size,
                              hipStream_t stream){
  const float* emb_f  = (const float*)d_in[0];
  const float* mask   = (const float*)d_in[1];
  const float* Wqg    = (const float*)d_in[2];
  const float* Wqs    = (const float*)d_in[3];
  const float* Wqt    = (const float*)d_in[4];
  const float* Wql    = (const float*)d_in[5];
  const float* Wk     = (const float*)d_in[6];
  const float* Wv     = (const float*)d_in[7];
  const float* Wc     = (const float*)d_in[8];
  const float* bc     = (const float*)d_in[9];
  const int*   srcn   = (const int*)d_in[10];
  const int*   tgtn   = (const int*)d_in[11];
  const int*   lastn  = (const int*)d_in[12];
  float* out = (float*)d_out;

  char* ws = (char*)d_ws;
  bf16*  emb_b = (bf16*)(ws + OFF_EMB);
  bf16*  Kb    = (bf16*)(ws + OFF_K);
  bf16*  Vt    = (bf16*)(ws + OFF_VT);
  bf16*  wkb   = (bf16*)(ws + OFF_WK);
  bf16*  wvb   = (bf16*)(ws + OFF_WV);
  bf16*  wqlb  = (bf16*)(ws + OFF_WQL);
  bf16*  wcb   = (bf16*)(ws + OFF_WC);
  float* gmean = (float*)(ws + OFF_MEAN);
  float* qbase = (float*)(ws + OFF_QB);
  bf16*  Qb    = (bf16*)(ws + OFF_Q);
  float* mh    = (float*)(ws + OFF_MH);
  bf16*  FQ    = (bf16*)(ws + OFF_FQ);
  float* rs    = (float*)(ws + OFF_RS);
  float* rsp   = (float*)(ws + OFF_RSP);
  unsigned char* mpk = (unsigned char*)(ws + OFF_MP);

  k_cvt_emb<<<64000, 256, 0, stream>>>(emb_f, emb_b);
  k_cvt_w  <<<256,   256, 0, stream>>>(Wk, Wv, Wql, Wc, wkb, wvb, wqlb, wcb);
  k_mpack  <<<8000,  256, 0, stream>>>(mask, mpk);
  k_mean   <<<32,    256, 0, stream>>>(emb_f, gmean);
  k_qbase  <<<32,    256, 0, stream>>>(gmean, emb_f, Wqg, Wqs, Wqt, srcn, tgtn, qbase);
  k_kproj  <<<1000,  256, 0, stream>>>(emb_b, wkb, Kb);
  k_vproj  <<<1000,  256, 0, stream>>>(emb_b, wvb, Vt);
  k_qlast  <<<64,    256, 0, stream>>>(emb_b, wqlb, qbase, lastn, Qb);
  k_attn   <<<1024,  128, 0, stream>>>(Qb, Kb, Vt, mpk, mh);
  k_fq     <<<64,    256, 0, stream>>>(mh, wcb, bc, FQ);
  k_logits <<<320,   256, 0, stream>>>(FQ, emb_b, mpk, out, rsp);
  k_sumpart<<<32,    256, 0, stream>>>(rsp, rs);
  k_norm   <<<64000, 256, 0, stream>>>(out, rs);
}
